// concat_mvPAM_Module_mask_cascade_13314398618067
// MI455X (gfx1250) — compile-verified, matrix operand order checked by reading, not run on hardware
//
#include <hip/hip_runtime.h>

#define B_ 2
#define C_ 512
#define D_ 64
#define N_ 4096
#define CAT_C 2048   // out0 | out1 | out2 | x  (4*C)

typedef __attribute__((ext_vector_type(16))) __bf16 v16bf;
typedef __attribute__((ext_vector_type(8)))  float  v8f;

union BfVec {
  v16bf v;
  uint4 q[2];
  unsigned short u[16];
};

__device__ inline unsigned short f2bf(float f) {
  unsigned int u = __float_as_uint(f);
  u += 0x7FFFu + ((u >> 16) & 1u);   // round-to-nearest-even
  return (unsigned short)(u >> 16);
}

__device__ inline v8f vzero8() {
  v8f z = {0.f, 0.f, 0.f, 0.f, 0.f, 0.f, 0.f, 0.f};
  return z;
}

__device__ inline v8f wmma_bf16(v16bf a, v16bf b, v8f c) {
  // D(f32 16x16) = A(bf16 16x32) * B(bf16 32x16) + C
  return __builtin_amdgcn_wmma_f32_16x16x32_bf16(false, a, false, b, (short)0, c,
                                                 false, false);
}

// ---------------- elementwise f32 -> bf16 ----------------
__global__ void cvt_bf16(const float* __restrict__ s, unsigned short* __restrict__ d,
                         int n) {
  int i = blockIdx.x * blockDim.x + threadIdx.x;
  if (i < n) d[i] = f2bf(s[i]);
}

// x[b][c][n] (f32) -> cat[b][n][1536+c] (bf16)   (also the B-operand for projections)
__global__ void x_to_cat(const float* __restrict__ x, unsigned short* __restrict__ cat) {
  int i = blockIdx.x * blockDim.x + threadIdx.x;
  if (i >= B_ * C_ * N_) return;
  int n = i % N_;
  int c = (i / N_) % C_;
  int b = i / (N_ * C_);
  cat[(size_t)(b * N_ + n) * CAT_C + 3 * C_ + c] = f2bf(x[i]);
}

// ---------------- generic bf16 WMMA GEMM ----------------
// Y[co][n] = sum_ci W[co][ci] * Xt[n][ci]      (Xt is position-major, row stride ldx)
// MODE 0: bf16 out, layout [b][n][cototal], + bias      (Q / K)
// MODE 1: bf16 out, layout [b][co][N], + bias           (V)
// MODE 2: f32  out, layout [b][co][N], BN + ReLU        (bottleneck -> d_out)
template <int MODE>
__global__ __launch_bounds__(256) void gemm_wmma(
    const unsigned short* __restrict__ W, const unsigned short* __restrict__ Xt,
    size_t xt_bstride, int ci, int ldx, void* __restrict__ out, int cototal,
    const float* __restrict__ bias, const float* __restrict__ bn_g,
    const float* __restrict__ bn_b, const float* __restrict__ bn_m,
    const float* __restrict__ bn_v) {
  const int b    = blockIdx.z;
  const int lane = threadIdx.x & 31;
  const int wave = threadIdx.x >> 5;
  const int nl   = lane & 15;
  const int half = lane >> 4;
  const int kb   = half * 8;                 // A/B k-pattern base per half-wave
  const int n0   = blockIdx.x * 64 + (wave & 3) * 16;
  const int co0  = blockIdx.y * 32 + (wave >> 2) * 16;

  const unsigned short* wr = W + (size_t)(co0 + nl) * ci;                 // A row (M=nl)
  const unsigned short* xr = Xt + (size_t)b * xt_bstride + (size_t)(n0 + nl) * ldx;

  v8f acc = vzero8();
  for (int k = 0; k < ci; k += 32) {
    BfVec a, xx;
    a.q[0]  = *(const uint4*)(wr + k + kb);
    a.q[1]  = *(const uint4*)(wr + k + 16 + kb);
    xx.q[0] = *(const uint4*)(xr + k + kb);
    xx.q[1] = *(const uint4*)(xr + k + 16 + kb);
    __builtin_prefetch(xr + k + 512, 0, 1);
    acc = wmma_bf16(a.v, xx.v, acc);
  }

  const int n = n0 + nl;  // D-tile column for this lane
#pragma unroll
  for (int r = 0; r < 8; ++r) {
    const int co = co0 + r + 8 * half;       // D-tile row
    float y = acc[r];
    if (MODE == 0) {
      y += bias[co];
      ((unsigned short*)out)[(size_t)(b * N_ + n) * cototal + co] = f2bf(y);
    } else if (MODE == 1) {
      y += bias[co];
      ((unsigned short*)out)[(size_t)(b * cototal + co) * N_ + n] = f2bf(y);
    } else {
      float inv = bn_g[co] * rsqrtf(bn_v[co] + 1e-5f);
      y = (y - bn_m[co]) * inv + bn_b[co];
      ((float*)out)[(size_t)(b * cototal + co) * N_ + n] = fmaxf(y, 0.f);
    }
  }
}

// ---------------- fused 3-mask flash attention ----------------
// Per workgroup: 16 attention rows (n0..n0+15), 8 waves x 64 channels of V/O.
// S^T = K * Q^T so that softmaxed P lands directly in A-operand layout for P*V.
__global__ __launch_bounds__(256) void flash3_wmma(
    const unsigned short* __restrict__ Qb, const unsigned short* __restrict__ Kb,
    const unsigned short* __restrict__ Vb, unsigned short* __restrict__ cat) {
  const int b    = blockIdx.y;
  const int n0   = blockIdx.x * 16;
  const int lane = threadIdx.x & 31;
  const int wave = threadIdx.x >> 5;
  const int nl   = lane & 15;
  const int half = lane >> 4;
  const int kb   = half * 8;
  const int nrow = n0 + nl;       // this lane's attention row (S^T column)
  const int c0   = wave * 64;     // this wave's channel slice

  // Q^T B-operand: lane holds column n=nrow, k=d pattern; resident across loop
  BfVec qB0, qB1;
  {
    const unsigned short* qp = Qb + (size_t)(b * N_ + nrow) * D_;
    qB0.q[0] = *(const uint4*)(qp + kb);
    qB0.q[1] = *(const uint4*)(qp + 16 + kb);
    qB1.q[0] = *(const uint4*)(qp + 32 + kb);
    qB1.q[1] = *(const uint4*)(qp + 48 + kb);
  }

  float mrun[3] = {-3.0e38f, -3.0e38f, -3.0e38f};
  float lrun[3] = {0.f, 0.f, 0.f};
  v8f acc[3][4];
#pragma unroll
  for (int i = 0; i < 3; ++i)
#pragma unroll
    for (int t = 0; t < 4; ++t) acc[i][t] = vzero8();

  for (int j = 0; j < N_ / 32; ++j) {
    const int m0 = j * 32;
    // band-block culling (exp(-1e4) == 0 in f32, so skipping == reference)
    const bool in1 = (m0 <= n0 + 15 + 1023) && (m0 + 31 >= n0 - 1023);
    const bool in2 = (m0 <= n0 + 15 + 255) && (m0 + 31 >= n0 - 255);

    // S^T[m, n] = sum_d K[m,d] * Q[n,d]  — two 16-row m tiles, K-dim 64 = 2 steps
    v8f S[2];
#pragma unroll
    for (int t = 0; t < 2; ++t) {
      const unsigned short* kp = Kb + (size_t)(b * N_ + m0 + 16 * t + nl) * D_;
      BfVec a0, a1;
      a0.q[0] = *(const uint4*)(kp + kb);
      a0.q[1] = *(const uint4*)(kp + 16 + kb);
      a1.q[0] = *(const uint4*)(kp + 32 + kb);
      a1.q[1] = *(const uint4*)(kp + 48 + kb);
      v8f s = vzero8();
      s = wmma_bf16(a0.v, qB0.v, s);
      s = wmma_bf16(a1.v, qB1.v, s);
      S[t] = s;
    }

    // V' B-operands (k=m, col=c); contiguous b128 loads from [c][m] layout
    BfVec vB[4];
#pragma unroll
    for (int t = 0; t < 4; ++t) {
      const unsigned short* vp = Vb + (size_t)(b * C_ + c0 + 16 * t + nl) * N_ + m0;
      vB[t].q[0] = *(const uint4*)(vp + kb);
      vB[t].q[1] = *(const uint4*)(vp + 16 + kb);
    }

    // this lane's 16 scores all belong to row nrow; m index per slot matches
    // the A-operand k-pattern of the upcoming P*V WMMA exactly.
    float sv[16];
    int dm[16];
#pragma unroll
    for (int r = 0; r < 8; ++r) {
      sv[r]     = S[0][r];
      sv[8 + r] = S[1][r];
      dm[r]     = abs(nrow - (m0 + kb + r));
      dm[8 + r] = abs(nrow - (m0 + 16 + kb + r));
    }

#pragma unroll
    for (int i = 0; i < 3; ++i) {
      if (i == 1 && !in1) continue;
      if (i == 2 && !in2) continue;
      const int band = (i == 0) ? (1 << 30) : ((i == 1) ? 1024 : 256);

      float bm = -3.0e38f;
#pragma unroll
      for (int e = 0; e < 16; ++e)
        if (dm[e] < band) bm = fmaxf(bm, sv[e]);
      bm = fmaxf(bm, __shfl_xor(bm, 16, 32));   // join lane pair sharing this row

      const float mn    = fmaxf(mrun[i], bm);
      const float scale = __expf(mrun[i] - mn);
      float ps = 0.f;
      BfVec P;
#pragma unroll
      for (int e = 0; e < 16; ++e) {
        float p = (dm[e] < band) ? __expf(sv[e] - mn) : 0.f;
        ps += p;
        P.u[e] = f2bf(p);
      }
      ps += __shfl_xor(ps, 16, 32);
      lrun[i] = lrun[i] * scale + ps;
      mrun[i] = mn;

      // rescale accumulators: acc row n' = n0 + r + 8*half, stats live in lane n'%16
#pragma unroll
      for (int r = 0; r < 8; ++r) {
        float fr = __shfl(scale, r + kb, 32);
#pragma unroll
        for (int t = 0; t < 4; ++t) acc[i][t][r] *= fr;
      }
#pragma unroll
      for (int t = 0; t < 4; ++t) acc[i][t] = wmma_bf16(P.v, vB[t].v, acc[i][t]);
    }
  }

  // normalize and emit bf16 into cat[b][n][i*512 + c]
#pragma unroll
  for (int i = 0; i < 3; ++i) {
#pragma unroll
    for (int r = 0; r < 8; ++r) {
      float inv = 1.f / __shfl(lrun[i], r + kb, 32);
      const int n = n0 + r + 8 * half;
#pragma unroll
      for (int t = 0; t < 4; ++t) {
        cat[(size_t)(b * N_ + n) * CAT_C + i * C_ + c0 + 16 * t + nl] =
            f2bf(acc[i][t][r] * inv);
      }
    }
  }
}

// ---------------- host ----------------
extern "C" void kernel_launch(void* const* d_in, const int* in_sizes, int n_in,
                              void* d_out, int out_size, void* d_ws, size_t ws_size,
                              hipStream_t stream) {
  const float* x    = (const float*)d_in[0];
  const float* wq   = (const float*)d_in[1];
  const float* bq   = (const float*)d_in[2];
  const float* wk   = (const float*)d_in[3];
  const float* bk   = (const float*)d_in[4];
  const float* wv   = (const float*)d_in[5];
  const float* bv   = (const float*)d_in[6];
  const float* wbot = (const float*)d_in[7];
  const float* g    = (const float*)d_in[8];
  const float* be   = (const float*)d_in[9];
  const float* mu   = (const float*)d_in[10];
  const float* var  = (const float*)d_in[11];
  // d_in[12], d_in[13] (mask1/mask2) unused: band structure computed from indices.
  float* out = (float*)d_out;

  char* ws = (char*)d_ws;
  size_t off = 0;
  unsigned short* CAT = (unsigned short*)(ws + off); off += (size_t)B_ * N_ * CAT_C * 2;
  unsigned short* Qb  = (unsigned short*)(ws + off); off += (size_t)B_ * N_ * D_ * 2;
  unsigned short* Kb  = (unsigned short*)(ws + off); off += (size_t)B_ * N_ * D_ * 2;
  unsigned short* Vb  = (unsigned short*)(ws + off); off += (size_t)B_ * C_ * N_ * 2;
  unsigned short* WQ  = (unsigned short*)(ws + off); off += (size_t)D_ * C_ * 2;
  unsigned short* WK  = (unsigned short*)(ws + off); off += (size_t)D_ * C_ * 2;
  unsigned short* WV  = (unsigned short*)(ws + off); off += (size_t)C_ * C_ * 2;
  unsigned short* WB  = (unsigned short*)(ws + off); off += (size_t)C_ * CAT_C * 2;

  auto cdiv = [](int a, int b) { return (a + b - 1) / b; };

  cvt_bf16<<<cdiv(D_ * C_, 256), 256, 0, stream>>>(wq, WQ, D_ * C_);
  cvt_bf16<<<cdiv(D_ * C_, 256), 256, 0, stream>>>(wk, WK, D_ * C_);
  cvt_bf16<<<cdiv(C_ * C_, 256), 256, 0, stream>>>(wv, WV, C_ * C_);
  cvt_bf16<<<cdiv(C_ * CAT_C, 256), 256, 0, stream>>>(wbot, WB, C_ * CAT_C);
  x_to_cat<<<cdiv(B_ * C_ * N_, 256), 256, 0, stream>>>(x, CAT);

  const unsigned short* XT = CAT + 3 * C_;   // x channels live in cat columns 1536..2047
  const size_t bstride = (size_t)N_ * CAT_C;

  gemm_wmma<0><<<dim3(N_ / 64, D_ / 32, B_), 256, 0, stream>>>(
      WQ, XT, bstride, C_, CAT_C, Qb, D_, bq, nullptr, nullptr, nullptr, nullptr);
  gemm_wmma<0><<<dim3(N_ / 64, D_ / 32, B_), 256, 0, stream>>>(
      WK, XT, bstride, C_, CAT_C, Kb, D_, bk, nullptr, nullptr, nullptr, nullptr);
  gemm_wmma<1><<<dim3(N_ / 64, C_ / 32, B_), 256, 0, stream>>>(
      WV, XT, bstride, C_, CAT_C, Vb, C_, bv, nullptr, nullptr, nullptr, nullptr);

  flash3_wmma<<<dim3(N_ / 16, B_), 256, 0, stream>>>(Qb, Kb, Vb, CAT);

  gemm_wmma<2><<<dim3(N_ / 64, C_ / 32, B_), 256, 0, stream>>>(
      WB, CAT, bstride, CAT_C, CAT_C, out, C_, nullptr, g, be, mu, var);
}
